// MultiheadAttention_34282428957091
// MI455X (gfx1250) — matrix operand order repaired and checked by reading, not compiled, not run on hardware
//
#include <hip/hip_runtime.h>
#include <hip/hip_bf16.h>

// ---------------------------------------------------------------------------
// Types for CDNA5 WMMA (wave32): A/B = 16 x bf16 (8 VGPRs), C/D = 8 x f32.
// ---------------------------------------------------------------------------
typedef __attribute__((ext_vector_type(16))) __bf16 v16bf;
typedef __attribute__((ext_vector_type(8)))  float  v8f;
typedef unsigned int v4u __attribute__((ext_vector_type(4), aligned(16)));

union Frag { v16bf v; v4u u[2]; };

#define BATCH  2
#define SLEN   2048
#define HEADS  8
#define DKV    64
#define DMODEL 512
#define NQKV   1536          // HEADS * (2*DK + DV)
#define MTOT   (BATCH * SLEN)

__device__ __forceinline__ unsigned short f2bf(float f) {
  unsigned int u = __float_as_uint(f);
  u += 0x7FFFu + ((u >> 16) & 1u);          // round-to-nearest-even
  return (unsigned short)(u >> 16);
}

__device__ __forceinline__ v8f zero8() {
  v8f z = {0.f, 0.f, 0.f, 0.f, 0.f, 0.f, 0.f, 0.f};
  return z;
}

__device__ __forceinline__ v8f wmma_bf16(const Frag& a, const Frag& b, v8f c) {
  return __builtin_amdgcn_wmma_f32_16x16x32_bf16(
      /*neg_a=*/false, a.v, /*neg_b=*/false, b.v,
      /*c_mod=*/(short)0, c, /*reuse_a=*/false, /*reuse_b=*/false);
}

// ---------------------------------------------------------------------------
// Kernel 1: convert x -> bf16; transpose+convert w_qkv, w_out -> [N][K] bf16.
// ---------------------------------------------------------------------------
__global__ void __launch_bounds__(256) prep_kernel(
    const float* __restrict__ x, const float* __restrict__ wqkv,
    const float* __restrict__ wout,
    unsigned short* __restrict__ xb,
    unsigned short* __restrict__ wqkvT,
    unsigned short* __restrict__ woutT)
{
  const int NX  = MTOT * DMODEL;      // 2097152
  const int NW1 = DMODEL * NQKV;      // 786432
  const int NW2 = DMODEL * DMODEL;    // 262144
  int idx = blockIdx.x * 256 + threadIdx.x;
  if (idx < NX) {
    xb[idx] = f2bf(x[idx]);
  } else if (idx < NX + NW1) {
    int j = idx - NX;
    int r = j / NQKV, c = j - r * NQKV;
    wqkvT[c * DMODEL + r] = f2bf(wqkv[j]);
  } else if (idx < NX + NW1 + NW2) {
    int j = idx - NX - NW1;
    int r = j / DMODEL, c = j - r * DMODEL;
    woutT[c * DMODEL + r] = f2bf(wout[j]);
  }
}

// ---------------------------------------------------------------------------
// Kernel 2/4: bf16 WMMA GEMM, M x 512 x N. Block tile 128x64, 8 waves,
// 32x32 per wave. A: [M][512] bf16 row-major, Bt: [N][512] bf16 (B^T).
// MODE 0: scatter into Q (scaled), K, V^T head-major buffers.
// MODE 1: plain f32 store to Cout [M][512].
// ---------------------------------------------------------------------------
template<int MODE>
__global__ void __launch_bounds__(256) gemm512_kernel(
    const unsigned short* __restrict__ A,
    const unsigned short* __restrict__ Bt,
    unsigned short* __restrict__ Qb,
    unsigned short* __restrict__ Kb,
    unsigned short* __restrict__ Vtb,
    float* __restrict__ Cout)
{
  __shared__ alignas(16) unsigned short As[128][40];  // 32 cols + pad (80B rows)
  __shared__ alignas(16) unsigned short Bs[64][40];

  const int tid   = threadIdx.x;
  const int lane  = tid & 31;
  const int wave  = tid >> 5;
  const int half  = lane >> 4;
  const int l16   = lane & 15;
  const int waveM = wave >> 1;                // 0..3
  const int waveN = wave & 1;                 // 0..1
  const int mBlock = blockIdx.x * 128;
  const int nBlock = blockIdx.y * 64;

  v8f acc[4];
#pragma unroll
  for (int i = 0; i < 4; ++i) acc[i] = zero8();

  const int arow = tid >> 1;                  // 0..127, 16 bf16 each
  const int acol = (tid & 1) * 16;
  const int brow = tid >> 2;                  // 0..63, 8 bf16 each
  const int bcol = (tid & 3) * 8;
  const unsigned short* aG = A  + (size_t)(mBlock + arow) * DMODEL + acol;
  const unsigned short* bG = Bt + (size_t)(nBlock + brow) * DMODEL + bcol;

  for (int k0 = 0; k0 < DMODEL; k0 += 32) {
    *(v4u*)&As[arow][acol]     = *(const v4u*)(aG + k0);
    *(v4u*)&As[arow][acol + 8] = *(const v4u*)(aG + k0 + 8);
    *(v4u*)&Bs[brow][bcol]     = *(const v4u*)(bG + k0);
    __syncthreads();

    Frag a0, a1, b0, b1;
    const int ar = waveM * 32 + l16;
    // A-fragment: lane row = l16, K = half*16 .. half*16+15 (contiguous)
    a0.u[0] = *(const v4u*)&As[ar     ][half * 16];
    a0.u[1] = *(const v4u*)&As[ar     ][half * 16 + 8];
    a1.u[0] = *(const v4u*)&As[ar + 16][half * 16];
    a1.u[1] = *(const v4u*)&As[ar + 16][half * 16 + 8];
    const int br = waveN * 32 + l16;
    // B-fragment: lane col = l16, K = half*16 .. half*16+15 (contiguous)
    b0.u[0] = *(const v4u*)&Bs[br     ][half * 16];
    b0.u[1] = *(const v4u*)&Bs[br     ][half * 16 + 8];
    b1.u[0] = *(const v4u*)&Bs[br + 16][half * 16];
    b1.u[1] = *(const v4u*)&Bs[br + 16][half * 16 + 8];

    acc[0] = wmma_bf16(a0, b0, acc[0]);
    acc[1] = wmma_bf16(a0, b1, acc[1]);
    acc[2] = wmma_bf16(a1, b0, acc[2]);
    acc[3] = wmma_bf16(a1, b1, acc[3]);
    __syncthreads();
  }

  const int mWave = mBlock + waveM * 32;
  const int nWave = nBlock + waveN * 32;
#pragma unroll
  for (int mi = 0; mi < 2; ++mi) {
#pragma unroll
    for (int ni = 0; ni < 2; ++ni) {
      v8f c = acc[mi * 2 + ni];
      const int nTile = nWave + ni * 16;
      if (MODE == 0) {
        const int h  = nTile / 192;           // uniform per 16-col tile
        const int r0 = nTile % 192;
#pragma unroll
        for (int i = 0; i < 8; ++i) {
          const int Mrow = mWave + mi * 16 + i + 8 * half;  // C row = i + 8*half
          const int bb = Mrow >> 11;
          const int ss = Mrow & (SLEN - 1);
          if (r0 < 64) {          // Q, pre-scaled by dk^-0.5 = 0.125
            Qb[((size_t)(bb * HEADS + h) * SLEN + ss) * DKV + r0 + l16] =
                f2bf(c[i] * 0.125f);
          } else if (r0 < 128) {  // K
            Kb[((size_t)(bb * HEADS + h) * SLEN + ss) * DKV + (r0 - 64) + l16] =
                f2bf(c[i]);
          } else {                // V stored transposed: [B,H,dv,S]
            Vtb[((size_t)(bb * HEADS + h) * DKV + (r0 - 128) + l16) * SLEN + ss] =
                f2bf(c[i]);
          }
        }
      } else {
#pragma unroll
        for (int i = 0; i < 8; ++i) {
          const int Mrow = mWave + mi * 16 + i + 8 * half;
          Cout[(size_t)Mrow * DMODEL + nTile + l16] = c[i];
        }
      }
    }
  }
}

// ---------------------------------------------------------------------------
// Kernel 3: flash attention. One wave = 16 query rows; block = 8 waves = 128
// rows of one (b, h). Keys processed 32 at a time: 4 WMMA for scores,
// online softmax (shfl_xor row-reductions within 16-lane halves),
// P -> LDS -> A-fragment, 4 WMMA for P*V (V is transposed in memory).
// ---------------------------------------------------------------------------
__global__ void __launch_bounds__(256) attn_kernel(
    const unsigned short* __restrict__ Qb,
    const unsigned short* __restrict__ Kb,
    const unsigned short* __restrict__ Vtb,
    const unsigned char*  __restrict__ maskp,
    unsigned short* __restrict__ attnOut)
{
  __shared__ alignas(16) unsigned short pbuf[8][16][40];  // per-wave P tile

  const int tid  = threadIdx.x;
  const int lane = tid & 31;
  const int wave = tid >> 5;
  const int half = lane >> 4;
  const int l16  = lane & 15;
  const int b = blockIdx.z;
  const int h = blockIdx.y;
  const int qbase = blockIdx.x * 128 + wave * 16;

  const unsigned short* Qh = Qb  + (size_t)(b * HEADS + h) * SLEN * DKV;
  const unsigned short* Kh = Kb  + (size_t)(b * HEADS + h) * SLEN * DKV;
  const unsigned short* Vh = Vtb + (size_t)(b * HEADS + h) * DKV * SLEN;
  const unsigned char*  mk = maskp + (size_t)b * SLEN;

  // Q A-fragments (dk 0..31 and 32..63), resident for the whole kernel.
  Frag qa0, qa1;
  {
    const unsigned short* qp = Qh + (size_t)(qbase + l16) * DKV;
    qa0.u[0] = *(const v4u*)(qp + half * 16);
    qa0.u[1] = *(const v4u*)(qp + half * 16 + 8);
    qa1.u[0] = *(const v4u*)(qp + 32 + half * 16);
    qa1.u[1] = *(const v4u*)(qp + 32 + half * 16 + 8);
  }

  v8f o[4];
#pragma unroll
  for (int j = 0; j < 4; ++j) o[j] = zero8();
  float mrow[8], lrow[8];
#pragma unroll
  for (int i = 0; i < 8; ++i) { mrow[i] = -__builtin_inff(); lrow[i] = 0.f; }

  unsigned short (*pb)[40] = pbuf[wave];

  for (int kt = 0; kt < SLEN; kt += 32) {
    // K B-fragments for two 16-key score tiles, dk split into 2 k-steps.
    const unsigned short* kp0 = Kh + (size_t)(kt + l16) * DKV;
    const unsigned short* kp1 = kp0 + 16 * DKV;
    Frag kf00, kf01, kf10, kf11;
    kf00.u[0] = *(const v4u*)(kp0 + half * 16);
    kf00.u[1] = *(const v4u*)(kp0 + half * 16 + 8);
    kf01.u[0] = *(const v4u*)(kp0 + 32 + half * 16);
    kf01.u[1] = *(const v4u*)(kp0 + 32 + half * 16 + 8);
    kf10.u[0] = *(const v4u*)(kp1 + half * 16);
    kf10.u[1] = *(const v4u*)(kp1 + half * 16 + 8);
    kf11.u[0] = *(const v4u*)(kp1 + 32 + half * 16);
    kf11.u[1] = *(const v4u*)(kp1 + 32 + half * 16 + 8);

    v8f s0 = wmma_bf16(qa0, kf00, zero8());
    s0     = wmma_bf16(qa1, kf01, s0);
    v8f s1 = wmma_bf16(qa0, kf10, zero8());
    s1     = wmma_bf16(qa1, kf11, s1);

    // Key-padding mask: lane column l16 = key index within tile.
    if (mk[kt + l16]) {
#pragma unroll
      for (int i = 0; i < 8; ++i) s0[i] = -__builtin_inff();
    }
    if (mk[kt + 16 + l16]) {
#pragma unroll
      for (int i = 0; i < 8; ++i) s1[i] = -__builtin_inff();
    }

    float p0[8], p1[8];
#pragma unroll
    for (int i = 0; i < 8; ++i) {
      float v = fmaxf(s0[i], s1[i]);
      v = fmaxf(v, __shfl_xor(v, 1, 32));
      v = fmaxf(v, __shfl_xor(v, 2, 32));
      v = fmaxf(v, __shfl_xor(v, 4, 32));
      v = fmaxf(v, __shfl_xor(v, 8, 32));
      const float mn    = fmaxf(mrow[i], v);
      const float alpha = (mrow[i] == -__builtin_inff()) ? 0.f
                                                         : __expf(mrow[i] - mn);
      const float e0 = __expf(s0[i] - mn);
      const float e1 = __expf(s1[i] - mn);
      float rs = e0 + e1;
      rs += __shfl_xor(rs, 1, 32);
      rs += __shfl_xor(rs, 2, 32);
      rs += __shfl_xor(rs, 4, 32);
      rs += __shfl_xor(rs, 8, 32);
      lrow[i] = lrow[i] * alpha + rs;
      mrow[i] = mn;
      o[0][i] *= alpha; o[1][i] *= alpha; o[2][i] *= alpha; o[3][i] *= alpha;
      p0[i] = e0; p1[i] = e1;
    }

    // C-layout -> A-layout transpose of P through per-wave LDS tile.
#pragma unroll
    for (int i = 0; i < 8; ++i) {
      pb[i + 8 * half][l16]      = f2bf(p0[i]);
      pb[i + 8 * half][16 + l16] = f2bf(p1[i]);
    }
    asm volatile("s_wait_dscnt 0x0" ::: "memory");  // same-wave DS is in-order

    Frag pf;
    pf.u[0] = *(const v4u*)&pb[l16][half * 16];
    pf.u[1] = *(const v4u*)&pb[l16][half * 16 + 8];

    // P(16x32) * V(32x64): V^T rows are contiguous in keys.
#pragma unroll
    for (int j = 0; j < 4; ++j) {
      Frag vf;
      const unsigned short* vp = Vh + (size_t)(j * 16 + l16) * SLEN + kt;
      vf.u[0] = *(const v4u*)(vp + half * 16);
      vf.u[1] = *(const v4u*)(vp + half * 16 + 8);
      o[j] = wmma_bf16(pf, vf, o[j]);
    }
  }

  // Normalize and write attention output [B, S, H*DV] in bf16.
#pragma unroll
  for (int i = 0; i < 8; ++i) {
    const float inv = 1.0f / lrow[i];
    const int row = b * SLEN + qbase + i + 8 * half;
    unsigned short* op = attnOut + (size_t)row * DMODEL + h * DKV;
    op[     l16] = f2bf(o[0][i] * inv);
    op[16 + l16] = f2bf(o[1][i] * inv);
    op[32 + l16] = f2bf(o[2][i] * inv);
    op[48 + l16] = f2bf(o[3][i] * inv);
  }
}

// ---------------------------------------------------------------------------
// Host-side launcher (graph-capture safe: kernel launches only).
// ---------------------------------------------------------------------------
extern "C" void kernel_launch(void* const* d_in, const int* in_sizes, int n_in,
                              void* d_out, int out_size, void* d_ws, size_t ws_size,
                              hipStream_t stream) {
  (void)in_sizes; (void)n_in; (void)out_size; (void)ws_size;
  const float*         x    = (const float*)d_in[0];
  const unsigned char* mask = (const unsigned char*)d_in[1];   // bool mask
  const float*         wqkv = (const float*)d_in[2];
  const float*         wout = (const float*)d_in[3];
  float*               out  = (float*)d_out;

  char* ws = (char*)d_ws;
  // Workspace layout (bytes), 256-aligned offsets.
  unsigned short* xb    = (unsigned short*)(ws + 0);          //  4 MB  [4096][512]
  unsigned short* wqkvT = (unsigned short*)(ws +  4194304);   //  1.5MB [1536][512]
  unsigned short* woutT = (unsigned short*)(ws +  5767168);   //  0.5MB [512][512]
  unsigned short* Qb    = (unsigned short*)(ws +  6291456);   //  4 MB  [B,H,S,64]
  unsigned short* Kb    = (unsigned short*)(ws + 10485760);   //  4 MB  [B,H,S,64]
  unsigned short* Vtb   = (unsigned short*)(ws + 14680064);   //  4 MB  [B,H,64,S]
  unsigned short* attn  = (unsigned short*)(ws + 0);          // aliases xb (safe:
                                                              // xb dead after GEMM1)

  // 1) fp32 -> bf16 conversions (+ weight transposes): 3,145,728 elements.
  prep_kernel<<<12288, 256, 0, stream>>>(x, wqkv, wout, xb, wqkvT, woutT);

  // 2) QKV projection: [4096 x 512] x [512 x 1536] -> Q/K/V^T (bf16 WMMA).
  gemm512_kernel<0><<<dim3(MTOT / 128, NQKV / 64), 256, 0, stream>>>(
      xb, wqkvT, Qb, Kb, Vtb, nullptr);

  // 3) Flash attention per (b, h, 128-row q tile).
  attn_kernel<<<dim3(SLEN / 128, HEADS, BATCH), 256, 0, stream>>>(
      Qb, Kb, Vtb, mask, attn);

  // 4) Output projection: [4096 x 512] x [512 x 512] -> f32 out.
  gemm512_kernel<1><<<dim3(MTOT / 128, DMODEL / 64), 256, 0, stream>>>(
      attn, woutT, nullptr, nullptr, nullptr, out);
}
